// AutoCorrelation_86045374808479
// MI455X (gfx1250) — matrix operand order repaired and checked by reading, not compiled, not run on hardware
//
#include <hip/hip_runtime.h>
#include <hip/hip_bf16.h>

typedef __attribute__((ext_vector_type(16))) __bf16 bf16x16;
typedef __attribute__((ext_vector_type(8)))  __bf16 bf16x8;
typedef __attribute__((ext_vector_type(8)))  float  f32x8;
typedef __attribute__((ext_vector_type(4)))  int    v4i;

#define B_   16
#define L_   4096
#define D_   512      // H*E
#define TOPK 24
#define NCH  32       // number of 128-wide block-diagonal chains (L/128)
#define NBIN 255      // relative delays covered by one chain: [-127,127]

#if defined(__has_builtin)
#  if __has_builtin(__builtin_amdgcn_global_load_async_to_lds_b128)
#    define HAVE_ASYNC_LDS 1
#  endif
#endif
#ifndef HAVE_ASYNC_LDS
#  define HAVE_ASYNC_LDS 0
#endif

#if HAVE_ASYNC_LDS
#  if __has_builtin(__builtin_amdgcn_s_wait_asynccnt)
#    define WAIT_ASYNC(n) __builtin_amdgcn_s_wait_asynccnt(n)
#  else
#    define WAIT_ASYNC(n) asm volatile("s_wait_asynccnt %0" ::"n"(n) : "memory")
#  endif
#endif

// Address-space-qualified v4i pointers for the async-to-LDS builtin.
#define GPTR(p) ((__attribute__((address_space(1))) v4i*)(p))
#define LPTR(p) ((__attribute__((address_space(3))) v4i*)(p))

// ---------------------------------------------------------------------------
// Kernel 1: fp32 -> bf16 (RNE) pack.  (B,L,H,E) is already row-major (l, he),
// so Qb[b] / Kb[b] are L x 512 bf16 row-major matrices.
// ---------------------------------------------------------------------------
__device__ __forceinline__ unsigned int f2bf2(float x, float y) {
    union { float f; unsigned u; } a, b;
    a.f = x; b.f = y;
    unsigned lo = (a.u + 0x7FFFu + ((a.u >> 16) & 1u)) >> 16;
    unsigned hi = (b.u + 0x7FFFu + ((b.u >> 16) & 1u)) >> 16;
    return lo | (hi << 16);
}

__global__ void __launch_bounds__(256) k_convert(const float* __restrict__ src,
                                                 unsigned short* __restrict__ dst,
                                                 int n8) {
    int t = blockIdx.x * 256 + threadIdx.x;
    if (t >= n8) return;
    const float4* s4 = reinterpret_cast<const float4*>(src) + (size_t)t * 2;
    float4 a = s4[0], b = s4[1];
    uint4 r;
    r.x = f2bf2(a.x, a.y);
    r.y = f2bf2(a.z, a.w);
    r.z = f2bf2(b.x, b.y);
    r.w = f2bf2(b.z, b.w);
    reinterpret_cast<uint4*>(dst)[t] = r;
}

// ---------------------------------------------------------------------------
// Kernel 2: block-diagonal correlation GEMM with WMMA.
// Grid: (chain d = 0..31, batch b = 0..15), 256 threads = 8 waves.
// Chain d = the 128x128 super-tiles (R,C) with R = (C+d) mod 32.
// tau = 128*d + (row-col) is invariant along the chain, so each wave's 2x4
// grid of 16x16 f32 accumulators persists across all 32 super-tiles x 16
// K-steps; one LDS bin reduction at the very end.
// Staging: GLOBAL_LOAD_ASYNC_TO_LDS_B128 (ASYNCcnt), 4 buffers, prefetch
// depth 3, fully uniform pipeline: wait<=8 / barrier / issue(it+3) / 8 WMMA.
// ---------------------------------------------------------------------------
__global__ void __launch_bounds__(256) k_corr(const unsigned short* __restrict__ Qb,
                                              const unsigned short* __restrict__ Kb,
                                              float* __restrict__ corrPart) {
    __shared__ unsigned short sAB[4][2][128 * 32];  // 4 stages x (A,B) x 8 KB
    __shared__ float bins[NBIN];

    const int tid = threadIdx.x;
    const int d   = blockIdx.x;
    const int b   = blockIdx.y;
    const unsigned short* Qbase = Qb + (size_t)b * L_ * D_;
    const unsigned short* Kbase = Kb + (size_t)b * L_ * D_;

    if (tid < NBIN) bins[tid] = 0.0f;

    const int lane = tid & 31;
    const int w    = tid >> 5;
    const int aRow = w & 3;       // row-pair selector  (i0 = 2*aRow)
    const int qCol = w >> 2;      // col-quad selector  (j0 = 4*qCol)
    const int rsel = lane & 15;
    const int hsel = lane >> 4;

    f32x8 acc[2][4];
    const f32x8 zero = {0.f, 0.f, 0.f, 0.f, 0.f, 0.f, 0.f, 0.f};
#pragma unroll
    for (int i = 0; i < 2; ++i)
#pragma unroll
        for (int j = 0; j < 4; ++j) acc[i][j] = zero;

    // Cooperative staging roles: threads 0..127 -> A slab row, 128..255 -> B.
    const bool isA  = (tid < 128);
    const int  lrow = isA ? tid : (tid - 128);
    const int  NIT  = NCH * 16;                     // 512 (power of two)

#if HAVE_ASYNC_LDS
    auto issueAsync = [&](int stage) {
        int sb = stage & 3;
        int C  = stage >> 4;
        int kk = stage & 15;
        int R  = (C + d) & 31;
        const unsigned short* src =
            isA ? (Qbase + (size_t)(R * 128 + lrow) * D_ + kk * 32)
                : (Kbase + (size_t)(C * 128 + lrow) * D_ + kk * 32);
        unsigned short* dst = &sAB[sb][isA ? 0 : 1][lrow * 32];
#pragma unroll
        for (int s = 0; s < 4; ++s)
            __builtin_amdgcn_global_load_async_to_lds_b128(
                GPTR(src + 8 * s), LPTR(dst + 8 * s), 0, 0);
    };
#else
    auto stageSync = [&](int stage, int sb) {
        int C  = stage >> 4;
        int kk = stage & 15;
        int R  = (C + d) & 31;
        const unsigned short* src =
            isA ? (Qbase + (size_t)(R * 128 + lrow) * D_ + kk * 32)
                : (Kbase + (size_t)(C * 128 + lrow) * D_ + kk * 32);
        unsigned short* dst = &sAB[sb][isA ? 0 : 1][lrow * 32];
#pragma unroll
        for (int s = 0; s < 4; ++s)
            *reinterpret_cast<uint4*>(dst + 8 * s) =
                *reinterpret_cast<const uint4*>(src + 8 * s);
    };
#endif

    auto computeStage = [&](int sb) {
        const unsigned short* Abuf = &sAB[sb][0][0];
        const unsigned short* Bbuf = &sAB[sb][1][0];
        // A operand (16x32): lane = row rsel; K = 8*hsel..+7 and +16..+23.
        bf16x16 amat[2];
#pragma unroll
        for (int di = 0; di < 2; ++di) {
            int i = 2 * aRow + di;
            int base = (i * 16 + rsel) * 32 + hsel * 8;
            bf16x8 lo = *reinterpret_cast<const bf16x8*>(&Abuf[base]);
            bf16x8 hi = *reinterpret_cast<const bf16x8*>(&Abuf[base + 16]);
            amat[di] = __builtin_shufflevector(lo, hi, 0, 1, 2, 3, 4, 5, 6, 7,
                                               8, 9, 10, 11, 12, 13, 14, 15);
        }
        // B operand (32x16): lane = col rsel; K = 8*hsel..+7 and +16..+23.
        bf16x16 bmat[4];
#pragma unroll
        for (int dj = 0; dj < 4; ++dj) {
            int j = 4 * qCol + dj;
            int base = (j * 16 + rsel) * 32 + hsel * 8;
            bf16x8 lo = *reinterpret_cast<const bf16x8*>(&Bbuf[base]);
            bf16x8 hi = *reinterpret_cast<const bf16x8*>(&Bbuf[base + 16]);
            bmat[dj] = __builtin_shufflevector(lo, hi, 0, 1, 2, 3, 4, 5, 6, 7,
                                               8, 9, 10, 11, 12, 13, 14, 15);
        }
#pragma unroll
        for (int di = 0; di < 2; ++di)
#pragma unroll
            for (int dj = 0; dj < 4; ++dj)
                acc[di][dj] = __builtin_amdgcn_wmma_f32_16x16x32_bf16(
                    false, amat[di], false, bmat[dj], (short)0, acc[di][dj],
                    false, false);
    };

#if HAVE_ASYNC_LDS
    issueAsync(0);
    issueAsync(1);
    issueAsync(2);
#pragma unroll 1
    for (int it = 0; it < NIT; ++it) {
        WAIT_ASYNC(8);                    // stage `it` landed (in-order)
        __syncthreads();                  // visible to all; buf (it+3)&3 free
        issueAsync((it + 3) & (NIT - 1)); // tail wraps to harmless re-loads
        computeStage(it & 3);
    }
#else
#pragma unroll 1
    for (int it = 0; it < NIT; ++it) {
        __syncthreads();                  // previous compute done
        stageSync(it, 0);
        __syncthreads();
        computeStage(0);
    }
#endif

    // Register pre-reduction: merge accumulators with equal (di - dj).
    f32x8 ga[5];
    ga[0] = acc[0][3];
    ga[1] = acc[0][2] + acc[1][3];
    ga[2] = acc[0][1] + acc[1][2];
    ga[3] = acc[0][0] + acc[1][1];
    ga[4] = acc[1][0];

    __syncthreads();   // bins zeroed long ago; all compute done

    // C/D layout: lane (rsel,hsel), VGPR v -> (M = v + 8*hsel, N = rsel).
    const int G = 2 * aRow - 4 * qCol;
#pragma unroll
    for (int r = 0; r < 5; ++r) {
        int g = G + (r - 3);
#pragma unroll
        for (int v = 0; v < 8; ++v) {
            int M = v + 8 * hsel;
            int bin = 16 * g + (M - rsel) + 127;   // in [0, 254]
            atomicAdd(&bins[bin], ga[r][v]);
        }
    }
    __syncthreads();
    if (tid < NBIN)
        corrPart[((size_t)(b * NCH + d)) * NBIN + tid] = bins[tid];
}

// ---------------------------------------------------------------------------
// Kernel 3: combine the <=2 chains contributing to each tau; produce
// mean_value[b,tau] (= corr/512) and the batch mean meanB[tau].
// ---------------------------------------------------------------------------
__global__ void __launch_bounds__(256) k_reduce(const float* __restrict__ corrPart,
                                                float* __restrict__ mean_value,
                                                float* __restrict__ meanB) {
    int tau = blockIdx.x * 256 + threadIdx.x;   // 0..4095
    int e = tau >> 7;
    int r = tau & 127;
    float sum = 0.f;
    for (int b = 0; b < B_; ++b) {
        float v = corrPart[((size_t)(b * NCH + e)) * NBIN + (r + 127)];
        if (r > 0)
            v += corrPart[((size_t)(b * NCH + ((e + 1) & (NCH - 1)))) * NBIN + (r - 1)];
        v *= (1.0f / 512.0f);
        mean_value[b * L_ + tau] = v;
        sum += v;
    }
    meanB[tau] = sum * (1.0f / 16.0f);
}

// ---------------------------------------------------------------------------
// Kernel 4: top-24 of meanB (single block, LDS-resident, lowest-index ties).
// ---------------------------------------------------------------------------
__global__ void __launch_bounds__(256) k_topk(const float* __restrict__ meanB,
                                              int* __restrict__ outIdx) {
    __shared__ float sv[L_];
    __shared__ float rv[256];
    __shared__ int   ri[256];
    int tid = threadIdx.x;
    for (int i = tid; i < L_; i += 256) sv[i] = meanB[i];
    __syncthreads();
    for (int k = 0; k < TOPK; ++k) {
        float bv = -3.4e38f;
        int bi = 0x7fffffff;
        for (int i = tid; i < L_; i += 256) {
            float v = sv[i];
            if (v > bv) { bv = v; bi = i; }
        }
        rv[tid] = bv; ri[tid] = bi;
        __syncthreads();
        for (int s = 128; s > 0; s >>= 1) {
            if (tid < s) {
                float ov = rv[tid + s]; int oi = ri[tid + s];
                if (ov > rv[tid] || (ov == rv[tid] && oi < ri[tid])) {
                    rv[tid] = ov; ri[tid] = oi;
                }
            }
            __syncthreads();
        }
        if (tid == 0) { outIdx[k] = ri[0]; sv[ri[0]] = -3.4e38f; }
        __syncthreads();
    }
}

// ---------------------------------------------------------------------------
// Kernel 5: per-batch softmax over the selected delays.
// ---------------------------------------------------------------------------
__global__ void k_softmax(const float* __restrict__ mean_value,
                          const int* __restrict__ idx,
                          float* __restrict__ wts) {
    int b = threadIdx.x;
    if (b >= B_) return;
    float v[TOPK];
    float m = -3.4e38f;
    for (int i = 0; i < TOPK; ++i) {
        v[i] = mean_value[b * L_ + idx[i]];
        m = fmaxf(m, v[i]);
    }
    float s = 0.f;
    for (int i = 0; i < TOPK; ++i) { v[i] = __expf(v[i] - m); s += v[i]; }
    float inv = 1.0f / s;
    for (int i = 0; i < TOPK; ++i) wts[b * TOPK + i] = v[i] * inv;
}

// ---------------------------------------------------------------------------
// Kernel 6: out[b,l,:] = sum_i w[b,i] * v[b,(l+idx[i]) mod L, :]  (float4).
// ---------------------------------------------------------------------------
__global__ void __launch_bounds__(256) k_agg(const float* __restrict__ vals,
                                             const int* __restrict__ idx,
                                             const float* __restrict__ wts,
                                             float4* __restrict__ out) {
    __shared__ int   sIdx[TOPK];
    __shared__ float sW[TOPK];
    int gid = blockIdx.x * 256 + threadIdx.x;
    int b = gid >> 19;                       // 4096*128 float4 per batch
    if (threadIdx.x < TOPK) sIdx[threadIdx.x] = idx[threadIdx.x];
    if (threadIdx.x >= 32 && threadIdx.x < 32 + TOPK)
        sW[threadIdx.x - 32] = wts[b * TOPK + (threadIdx.x - 32)];
    __syncthreads();
    int rem = gid & ((1 << 19) - 1);
    int l = rem >> 7;
    int c = rem & 127;
    const float4* vb = reinterpret_cast<const float4*>(vals) + ((size_t)b << 19);
    float4 o = {0.f, 0.f, 0.f, 0.f};
#pragma unroll 4
    for (int i = 0; i < TOPK; ++i) {
        int ll = (l + sIdx[i]) & (L_ - 1);
        float4 x = vb[((size_t)ll << 7) + c];
        float wgt = sW[i];
        o.x += wgt * x.x; o.y += wgt * x.y; o.z += wgt * x.z; o.w += wgt * x.w;
    }
    out[gid] = o;
}

// ---------------------------------------------------------------------------
extern "C" void kernel_launch(void* const* d_in, const int* in_sizes, int n_in,
                              void* d_out, int out_size, void* d_ws, size_t ws_size,
                              hipStream_t stream) {
    const float* q = (const float*)d_in[0];
    const float* k = (const float*)d_in[1];
    const float* v = (const float*)d_in[2];

    const size_t NEL = (size_t)B_ * L_ * D_;          // 33,554,432 elements
    char* ws = (char*)d_ws;
    unsigned short* Qb = (unsigned short*)ws;          // NEL * 2 bytes
    unsigned short* Kb = Qb + NEL;                     // NEL * 2 bytes
    float* corrPart   = (float*)(ws + 2 * NEL * sizeof(unsigned short));
    float* mean_value = corrPart + (size_t)B_ * NCH * NBIN;
    float* meanB      = mean_value + (size_t)B_ * L_;
    int*   idx        = (int*)(meanB + L_);
    float* wts        = (float*)(idx + TOPK);

    const int n8 = (int)(NEL / 8);
    k_convert<<<n8 / 256, 256, 0, stream>>>(q, Qb, n8);
    k_convert<<<n8 / 256, 256, 0, stream>>>(k, Kb, n8);

    dim3 g2(NCH, B_);
    k_corr<<<g2, 256, 0, stream>>>(Qb, Kb, corrPart);

    k_reduce<<<L_ / 256, 256, 0, stream>>>(corrPart, mean_value, meanB);
    k_topk<<<1, 256, 0, stream>>>(meanB, idx);
    k_softmax<<<1, 32, 0, stream>>>(mean_value, idx, wts);

    k_agg<<<(B_ * L_ * 128) / 256, 256, 0, stream>>>(v, idx, wts, (float4*)d_out);
}
